// MultiHeadAttention_9328668967781
// MI455X (gfx1250) — matrix operand order repaired and checked by reading, not compiled, not run on hardware
//
#include <hip/hip_runtime.h>
#include <hip/hip_bf16.h>
#include <type_traits>

// MI455X / gfx1250, wave32. bf16 WMMA everywhere; async-to-LDS double-buffered
// flash attention with register-resident online softmax.
// Pipeline: 3x projection GEMM (Q,K normal / V transposed) -> fused causal
// flash-attention -> output GEMM.

typedef __attribute__((ext_vector_type(16))) __bf16 v16bf;
typedef __attribute__((ext_vector_type(8)))  float  v8f;

#define WMMA_BF16(a, b, c) \
  __builtin_amdgcn_wmma_f32_16x16x32_bf16(false, (a), false, (b), (short)0, (c), false, false)

static __device__ __forceinline__ unsigned short f2bf(float f) {
  union { float f; unsigned u; } v; v.f = f;
  unsigned r = v.u + 0x7FFFu + ((v.u >> 16) & 1u);   // round-to-nearest-even
  return (unsigned short)(r >> 16);
}

union Frag { uint4 u[2]; v16bf v; };   // 32B: 16x32 bf16 A-frag / 32x16 B-frag per lane

// ---- gfx1250 async global->LDS copy (ASYNCcnt-tracked, no VGPR data path) ----
static __device__ __forceinline__ void asy_b128(unsigned lds_off, const void* g) {
  asm volatile("global_load_async_to_lds_b128 %0, %1, off" :: "v"(lds_off), "v"(g) : "memory");
}
static __device__ __forceinline__ void wait_async0() {
  asm volatile("s_wait_asynccnt 0" ::: "memory");
}
static __device__ __forceinline__ unsigned lds_off32(const void* p) {
  // flat LDS aperture: addr[31:0] is the in-LDS byte offset (ISA 10.2)
  return (unsigned)(uintptr_t)p;
}

// ---------------------------------------------------------------------------
// GEMM:  C[8192 x 1024] = A[8192 x 1024] * W[1024 x 1024] (+ bias)
// MODE 0: A fp32, out bf16 (B,H,L,DH), bias        (Q, K projection)
// MODE 1: A fp32, out bf16 (B,H,DH,L) transposed,  (V projection; async-friendly)
// MODE 3: A bf16 ctx, out fp32 (B,L,D), no bias    (output projection)
// Block 256 thr (8 waves), tile 128x64, wave = 32x32 (2x2 WMMA tiles).
// ---------------------------------------------------------------------------
template<int MODE>
__global__ __launch_bounds__(256) void gemm_wmma(const void* __restrict__ Ap,
                                                 const float* __restrict__ W,
                                                 const float* __restrict__ bias,
                                                 void* __restrict__ Out)
{
  constexpr int K = 1024, N = 1024, L = 2048, H = 16, DH = 64;
  __shared__ __align__(16) unsigned short As[128][32];
  __shared__ __align__(16) unsigned short Wt[64][32];   // [n][k]

  const int tid  = threadIdx.x;
  const int ln   = tid & 31;
  const int w    = tid >> 5;
  const int half = ln >> 4;
  const int mrow = ln & 15;
  const int n0 = blockIdx.x * 64;
  const int m0 = blockIdx.y * 128;
  const int wm = (w >> 1) * 32;
  const int wn = (w & 1) * 32;

  v8f acc00 = {}, acc01 = {}, acc10 = {}, acc11 = {};

  const int arow = tid >> 3;
  const int acol = (tid & 7) * 4;
  const int wk   = tid >> 4;
  const int wnn  = (tid & 15) * 4;

  for (int ko = 0; ko < K; ko += 32) {
    if (MODE == 3) {
      const unsigned short* A = (const unsigned short*)Ap;
      for (int rr = 0; rr < 4; ++rr) {
        int r = arow + rr * 32;
        uint2 d = *(const uint2*)(A + (size_t)(m0 + r) * K + ko + acol);
        *(uint2*)&As[r][acol] = d;
      }
    } else {
      const float* A = (const float*)Ap;
      for (int rr = 0; rr < 4; ++rr) {
        int r = arow + rr * 32;
        float4 d = *(const float4*)(A + (size_t)(m0 + r) * K + ko + acol);
        unsigned p0 = (unsigned)f2bf(d.x) | ((unsigned)f2bf(d.y) << 16);
        unsigned p1 = (unsigned)f2bf(d.z) | ((unsigned)f2bf(d.w) << 16);
        *(uint2*)&As[r][acol] = make_uint2(p0, p1);
      }
    }
    for (int t = 0; t < 2; ++t) {
      int k = wk + t * 16;
      float4 d = *(const float4*)(W + (size_t)(ko + k) * N + n0 + wnn);
      Wt[wnn + 0][k] = f2bf(d.x);
      Wt[wnn + 1][k] = f2bf(d.y);
      Wt[wnn + 2][k] = f2bf(d.z);
      Wt[wnn + 3][k] = f2bf(d.w);
    }
    if (ko + 32 < K) {   // -> global_prefetch_b8
      __builtin_prefetch((const char*)Ap +
          ((size_t)(m0 + arow) * K + ko + 32 + acol) * (MODE == 3 ? 2 : 4), 0, 3);
      __builtin_prefetch(W + (size_t)(ko + 32 + wk) * N + n0 + wnn, 0, 3);
    }
    __syncthreads();

    Frag a0, a1, b0, b1;
    a0.u[0] = *(const uint4*)&As[wm + mrow][half * 16];
    a0.u[1] = *(const uint4*)&As[wm + mrow][half * 16 + 8];
    a1.u[0] = *(const uint4*)&As[wm + 16 + mrow][half * 16];
    a1.u[1] = *(const uint4*)&As[wm + 16 + mrow][half * 16 + 8];
    b0.u[0] = *(const uint4*)&Wt[wn + mrow][half * 16];
    b0.u[1] = *(const uint4*)&Wt[wn + mrow][half * 16 + 8];
    b1.u[0] = *(const uint4*)&Wt[wn + 16 + mrow][half * 16];
    b1.u[1] = *(const uint4*)&Wt[wn + 16 + mrow][half * 16 + 8];

    acc00 = WMMA_BF16(a0.v, b0.v, acc00);
    acc01 = WMMA_BF16(a0.v, b1.v, acc01);
    acc10 = WMMA_BF16(a1.v, b0.v, acc10);
    acc11 = WMMA_BF16(a1.v, b1.v, acc11);
    __syncthreads();
  }

  const int ng0 = n0 + wn + mrow;
  const int ng1 = ng0 + 16;
  float bv0 = 0.f, bv1 = 0.f;
  if (MODE != 3) { bv0 = bias[ng0]; bv1 = bias[ng1]; }

  for (int i = 0; i < 8; ++i) {
    int mg0 = m0 + wm + i + 8 * half;
    int mg1 = mg0 + 16;
    float v00 = acc00[i] + bv0, v01 = acc01[i] + bv1;
    float v10 = acc10[i] + bv0, v11 = acc11[i] + bv1;
    if (MODE == 3) {
      float* O = (float*)Out;
      O[(size_t)mg0 * N + ng0] = v00;
      O[(size_t)mg0 * N + ng1] = v01;
      O[(size_t)mg1 * N + ng0] = v10;
      O[(size_t)mg1 * N + ng1] = v11;
    } else {
      unsigned short* O = (unsigned short*)Out;
      auto st = [&](int mg, int ng, float v) {
        int bb = mg >> 11, l = mg & (L - 1), h = ng >> 6, dh = ng & (DH - 1);
        if (MODE == 1)      // transposed (B,H,DH,L)
          O[(((size_t)(bb * H + h) * DH + dh) * L) + l] = f2bf(v);
        else                // (B,H,L,DH)
          O[(((size_t)(bb * H + h) * L + l) * DH) + dh] = f2bf(v);
      };
      st(mg0, ng0, v00); st(mg0, ng1, v01); st(mg1, ng0, v10); st(mg1, ng1, v11);
    }
  }
}

// ---------------------------------------------------------------------------
// Fused causal flash-attention, async double-buffered K/V, register softmax.
// Block 128 thr (4 waves); 64 queries/block; 64-key chunks, causal skip.
// Diagonal chunk handled by a separate compile-time-specialized body (no
// per-element branches around the attn_mask loads).
// ---------------------------------------------------------------------------
__global__ __launch_bounds__(128) void flash_attn(const unsigned short* __restrict__ Qb,
                                                  const unsigned short* __restrict__ Kb,
                                                  const unsigned short* __restrict__ Vt,
                                                  const float* __restrict__ amask,
                                                  const unsigned char* __restrict__ padd,
                                                  unsigned short* __restrict__ Ctx)
{
  constexpr int L = 2048, D = 1024, DH = 64;
  __shared__ __align__(16) unsigned short Qs[64][64];      // 8 KB
  __shared__ __align__(16) unsigned short Ks[2][64][64];   // 16 KB [key][dh]
  __shared__ __align__(16) unsigned short Vs[2][64][64];   // 16 KB [dh][key]
  __shared__ __align__(16) unsigned short Ps[64][64];      // 8 KB  probs (intra-wave only)
  __shared__ float PdF[2048];                              // <=8 KB padd row as float

  const int tid  = threadIdx.x;
  const int ln   = tid & 31, w = tid >> 5;
  const int half = ln >> 4, mrow = ln & 15;
  const int bh = blockIdx.y;
  const int b  = bh >> 4, h = bh & 15;
  const int Q0 = blockIdx.x * 64;
  const int m0 = w * 16;
  const int jn = blockIdx.x + 1;            // causal chunk skip

  const char* kgbase = (const char*)(Kb + (size_t)bh * L * DH);   // rows contiguous
  const char* vgbase = (const char*)(Vt + (size_t)bh * DH * L);   // dh-major, stride L

  // async fill of K/V chunk j into buffer j&1 (8 KB each, 4 b128 per thread)
  auto fillKV = [&](int j) {
    const int cur = j & 1, k0 = j * 64;
    const unsigned ka = lds_off32(&Ks[cur][0][0]);
    const unsigned va = lds_off32(&Vs[cur][0][0]);
    const char* kg = kgbase + (size_t)k0 * DH * 2;
    const char* vg = vgbase + (size_t)k0 * 2;
    for (int ii = 0; ii < 4; ++ii) {
      int p = tid + ii * 128;                 // 0..511 16B pieces
      unsigned off = (unsigned)p * 16u;
      asy_b128(ka + off, kg + off);           // K rows are contiguous
      int dh = p >> 3, kb8 = (p & 7) * 16;
      asy_b128(va + off, vg + (size_t)dh * (L * 2) + kb8);  // V^T rows strided
    }
  };

  { // async fill of the Q tile (contiguous 8 KB) + first K/V chunk
    const char* qg = (const char*)(Qb + ((size_t)bh * L + Q0) * DH);
    const unsigned qa = lds_off32(&Qs[0][0]);
    for (int ii = 0; ii < 4; ++ii) {
      unsigned off = (unsigned)(tid + ii * 128) * 16u;
      asy_b128(qa + off, qg + off);
    }
    fillKV(0);
  }
  // preload this batch row's padd mask (0/1) as float into LDS
  {
    const unsigned char* pdrow = padd + (size_t)b * L;
    for (int idx = tid; idx < jn * 64; idx += 128) PdF[idx] = (float)pdrow[idx];
  }
  wait_async0();
  __syncthreads();

  // Q fragments (hoisted: constant across chunks)
  Frag q0f, q1f;
  q0f.u[0] = *(const uint4*)&Qs[m0 + mrow][half * 16];
  q0f.u[1] = *(const uint4*)&Qs[m0 + mrow][half * 16 + 8];
  q1f.u[0] = *(const uint4*)&Qs[m0 + mrow][32 + half * 16];
  q1f.u[1] = *(const uint4*)&Qs[m0 + mrow][32 + half * 16 + 8];

  v8f accO[4] = { {}, {}, {}, {} };
  float rM[8], rL[8];
#pragma unroll
  for (int i = 0; i < 8; ++i) { rM[i] = -1e30f; rL[i] = 0.f; }

  // one chunk of work; DIAG is a compile-time constant at each call site
  auto chunk = [&](int j, auto diagc) {
    constexpr bool DIAG = decltype(diagc)::value;
    const int cur = j & 1, k0 = j * 64;
    const unsigned short (*Kc)[64] = Ks[cur];
    const unsigned short (*Vc)[64] = Vs[cur];

    // ---- S = Q K^T (8 WMMAs) ----
    v8f s[4];
#pragma unroll
    for (int nt = 0; nt < 4; ++nt) {
      Frag kf0, kf1;
      kf0.u[0] = *(const uint4*)&Kc[nt * 16 + mrow][half * 16];
      kf0.u[1] = *(const uint4*)&Kc[nt * 16 + mrow][half * 16 + 8];
      kf1.u[0] = *(const uint4*)&Kc[nt * 16 + mrow][32 + half * 16];
      kf1.u[1] = *(const uint4*)&Kc[nt * 16 + mrow][32 + half * 16 + 8];
      v8f z = {};
      z = WMMA_BF16(q0f.v, kf0.v, z);
      s[nt] = WMMA_BF16(q1f.v, kf1.v, z);
    }

    // ---- scale + padd (+ attn_mask, diagonal chunk only: batched loads) ----
    float pd[4];
#pragma unroll
    for (int nt = 0; nt < 4; ++nt) pd[nt] = PdF[k0 + nt * 16 + mrow];

    float mv[4][8];
    if constexpr (DIAG) {
      const float* mbase = amask + (size_t)(Q0 + m0 + 8 * half) * L + k0 + mrow;
#pragma unroll
      for (int nt = 0; nt < 4; ++nt)
#pragma unroll
        for (int i = 0; i < 8; ++i)
          mv[nt][i] = mbase[(size_t)i * L + nt * 16];
    }

#pragma unroll
    for (int nt = 0; nt < 4; ++nt)
#pragma unroll
      for (int i = 0; i < 8; ++i) {
        float v = s[nt][i] * 0.125f + pd[nt];
        if constexpr (DIAG) v += mv[nt][i];
        s[nt][i] = v;
      }

    // ---- register online softmax: rows live across 16-lane halves ----
#pragma unroll
    for (int i = 0; i < 8; ++i) {
      float mx = fmaxf(fmaxf(s[0][i], s[1][i]), fmaxf(s[2][i], s[3][i]));
#pragma unroll
      for (int d = 1; d < 16; d <<= 1) mx = fmaxf(mx, __shfl_xor(mx, d, 32));
      float mnew = fmaxf(rM[i], mx);
      float cf = __expf(rM[i] - mnew);
      rM[i] = mnew;
      float p0 = __expf(s[0][i] - mnew), p1 = __expf(s[1][i] - mnew);
      float p2 = __expf(s[2][i] - mnew), p3 = __expf(s[3][i] - mnew);
      s[0][i] = p0; s[1][i] = p1; s[2][i] = p2; s[3][i] = p3;
      float ps = (p0 + p1) + (p2 + p3);
#pragma unroll
      for (int d = 1; d < 16; d <<= 1) ps += __shfl_xor(ps, d, 32);
      rL[i] = rL[i] * cf + ps;
      accO[0][i] *= cf; accO[1][i] *= cf; accO[2][i] *= cf; accO[3][i] *= cf;
    }

    // ---- P tile -> LDS (bf16); intra-wave rows only, no barrier needed ----
#pragma unroll
    for (int nt = 0; nt < 4; ++nt)
#pragma unroll
      for (int i = 0; i < 8; ++i)
        Ps[m0 + i + 8 * half][nt * 16 + mrow] = f2bf(s[nt][i]);

    // ---- O += P V (8 WMMAs) ----
#pragma unroll
    for (int kt = 0; kt < 2; ++kt) {
      Frag pf;
      pf.u[0] = *(const uint4*)&Ps[m0 + mrow][kt * 32 + half * 16];
      pf.u[1] = *(const uint4*)&Ps[m0 + mrow][kt * 32 + half * 16 + 8];
#pragma unroll
      for (int nt = 0; nt < 4; ++nt) {
        Frag vf;
        vf.u[0] = *(const uint4*)&Vc[nt * 16 + mrow][kt * 32 + half * 16];
        vf.u[1] = *(const uint4*)&Vc[nt * 16 + mrow][kt * 32 + half * 16 + 8];
        accO[nt] = WMMA_BF16(pf.v, vf.v, accO[nt]);
      }
    }
  };

  // full (mask-free) chunks, double-buffered async prefetch of the next chunk
  for (int j = 0; j < jn - 1; ++j) {
    fillKV(j + 1);
    chunk(j, std::integral_constant<bool, false>{});
    wait_async0();        // next chunk's copy complete & all waves done with
    __syncthreads();      // the buffer it overwrote
  }
  // diagonal chunk (attn_mask applied, straight-line batched loads)
  chunk(jn - 1, std::integral_constant<bool, true>{});

  // ---- finalize: divide by row sum, write ctx (B,L,D) bf16 ----
  float li[8];
#pragma unroll
  for (int i = 0; i < 8; ++i) li[i] = 1.0f / rL[i];
#pragma unroll
  for (int nt = 0; nt < 4; ++nt) {
    int dh = nt * 16 + mrow;
#pragma unroll
    for (int i = 0; i < 8; ++i) {
      int q = Q0 + m0 + i + 8 * half;
      Ctx[((size_t)b * L + q) * D + h * DH + dh] = f2bf(accO[nt][i] * li[i]);
    }
  }
}

// ---------------------------------------------------------------------------
extern "C" void kernel_launch(void* const* d_in, const int* in_sizes, int n_in,
                              void* d_out, int out_size, void* d_ws, size_t ws_size,
                              hipStream_t stream)
{
  const float* Xq = (const float*)d_in[0];
  const float* Xk = (const float*)d_in[1];
  const float* Xv = (const float*)d_in[2];
  const float* amask = (const float*)d_in[3];
  const unsigned char* padd = (const unsigned char*)d_in[4];  // bool (B, L)
  const float* Wq = (const float*)d_in[5];
  const float* bq = (const float*)d_in[6];
  const float* Wk = (const float*)d_in[7];
  const float* bk = (const float*)d_in[8];
  const float* Wv = (const float*)d_in[9];
  const float* bv = (const float*)d_in[10];
  const float* Wo = (const float*)d_in[11];

  const size_t elems = (size_t)8192 * 1024;           // B*L*D
  unsigned short* Qb  = (unsigned short*)d_ws;        // bf16 (B,H,L,DH)
  unsigned short* Kb  = Qb + elems;                   // bf16 (B,H,L,DH)
  unsigned short* Vtb = Kb + elems;                   // bf16 (B,H,DH,L)  (transposed)
  unsigned short* Ctx = Vtb + elems;                  // bf16 (B,L,D)

  dim3 gg(16, 64);                                    // N/64 x M/128
  gemm_wmma<0><<<gg, 256, 0, stream>>>(Xq, Wq, bq, Qb);
  gemm_wmma<0><<<gg, 256, 0, stream>>>(Xk, Wk, bk, Kb);
  gemm_wmma<1><<<gg, 256, 0, stream>>>(Xv, Wv, bv, Vtb);
  flash_attn<<<dim3(32, 64), 128, 0, stream>>>(Qb, Kb, Vtb, amask, padd, Ctx);
  gemm_wmma<3><<<gg, 256, 0, stream>>>(Ctx, Wo, nullptr, d_out);
}
